// BiDAF_28982439313470
// MI455X (gfx1250) — matrix operand order repaired and checked by reading, not compiled, not run on hardware
//
#include <hip/hip_runtime.h>

// ---------------- problem constants ----------------
constexpr int NB = 64;     // batch
constexpr int NT = 2048;   // T
constexpr int NJ = 256;    // J
constexpr int ND = 200;    // D
constexpr int NG = 4 * ND; // g row length = 800

// ---------------- WMMA vector types ----------------
typedef __attribute__((ext_vector_type(16))) _Float16 v16h;
typedef __attribute__((ext_vector_type(8)))  _Float16 v8h;
typedef __attribute__((ext_vector_type(8)))  float    v8f;

// ---------------- LDS layout (bytes) ----------------
// strides in f16 elements, chosen 8-aligned (16B vector loads) and odd-ish
// multiples of 16B to spread LDS banks.
constexpr int LD_UH = 232;  // u row-major   : 256 x 232 f16 (cols >=200 zero)
constexpr int LD_UT = 264;  // u transposed  : 224 x 264 f16 (rows >=200 zero)
constexpr int LD_HW = 232;  // hw tile       :  64 x 232 f16 (cols >=200 zero)
constexpr int LD_P  = 264;  // P tile        :  64 x 264 f16
constexpr int LD_S  = 260;  // s tile (f32)  :  64 x 260 f32

constexpr int OFF_UH = 0;
constexpr int SZ_UH  = NJ * LD_UH * 2;          // 118784
constexpr int OFF_UT = OFF_UH + SZ_UH;          // 118784 (aliased by s after GEMM1)
constexpr int SZ_UT  = 224 * LD_UT * 2;         // 118272
constexpr int OFF_P  = OFF_UT + SZ_UT;          // 237056 (hw first, P after softmax)
constexpr int SZ_P   = 64 * LD_P * 2;           // 33792
constexpr int OFF_TU = OFF_P + SZ_P;            // 270848
constexpr int OFF_TH = OFF_TU + NJ * 4;         // 271872
constexpr int OFF_RS = OFF_TH + 64 * 4;         // 272128
constexpr int SMEM_TOTAL = OFF_RS + 64 * 4;     // 272384 bytes  (< 320KB/WGP)

// ---------------- WMMA fragment loaders (CDNA5 ISA 7.12.2, wave32) ----------------
// A (16x32 f16): lanes 0-15 -> M=lane, K in {0..7,16..23}; lanes 16-31 -> M=lane-16,
// K in {8..15,24..31}. Element 2v/2v+1 of v16h = lo/hi half of VGPR v.
__device__ __forceinline__ v16h frag_a(const _Float16* base, int ld) {
  const int lane = threadIdx.x & 31;
  const int m  = lane & 15;
  const int kq = (lane >> 4) << 3;                 // 0 or 8
  const _Float16* p0 = base + m * ld + kq;         // K = kq .. kq+7
  const _Float16* p1 = p0 + 16;                    // K = 16+kq .. 16+kq+7
  v8h lo = *(const v8h*)p0;
  v8h hi = *(const v8h*)p1;
  v16h a;
#pragma unroll
  for (int i = 0; i < 8; ++i) { a[i] = lo[i]; a[8 + i] = hi[i]; }
  return a;
}

// B (32x16 f16): lanes 0-15 -> N=lane, K in {0..7,16..23}; lanes 16-31 -> N=lane-16,
// K in {8..15,24..31}. base points at B[k0][n0] in row-major [K][N] storage.
__device__ __forceinline__ v16h frag_b(const _Float16* base, int ld) {
  const int lane = threadIdx.x & 31;
  const int n  = lane & 15;
  const int kq = (lane >> 4) << 3;
  v16h b;
#pragma unroll
  for (int i = 0; i < 8; ++i) {
    b[i]     = base[(kq + i) * ld + n];
    b[8 + i] = base[(16 + kq + i) * ld + n];
  }
  return b;
}

// ======================================================================
// Kernel 1: fused  s -> softmax_j -> c2q ; writes g[.,.,D:3D] and mrow
// grid (NT/64, NB), 256 threads (8 waves), SMEM_TOTAL dynamic LDS
// ======================================================================
__global__ __launch_bounds__(256)
void bidaf_attn_kernel(const float* __restrict__ h, const float* __restrict__ u,
                       const float* __restrict__ w_h, const float* __restrict__ b_h,
                       const float* __restrict__ w_u, const float* __restrict__ b_u,
                       const float* __restrict__ w_hu, const float* __restrict__ b_hu,
                       float* __restrict__ g, float* __restrict__ mrow_g)
{
  extern __shared__ char smem[];
  _Float16* uh = (_Float16*)(smem + OFF_UH);   // u row-major f16
  _Float16* uT = (_Float16*)(smem + OFF_UT);   // u transposed f16 (GEMM1 B operand)
  float*    sS = (float*)   (smem + OFF_UT);   // s tile f32 (aliases uT after GEMM1)
  _Float16* hw = (_Float16*)(smem + OFF_P);    // h*w_hu f16 (GEMM1 A operand)
  _Float16* P  = (_Float16*)(smem + OFF_P);    // softmax exps f16 (aliases hw)
  float* tu = (float*)(smem + OFF_TU);
  float* th = (float*)(smem + OFF_TH);
  float* rs = (float*)(smem + OFF_RS);

  const int tid  = threadIdx.x;
  const int lane = tid & 31;
  const int w    = tid >> 5;                   // wave id 0..7
  const int b    = blockIdx.y;
  const int t0   = blockIdx.x * 64;

  // zero all LDS: provides the K/N zero-padding for uh/uT/hw in one shot
  for (int i = tid; i < SMEM_TOTAL / 16; i += 256)
    ((int4*)smem)[i] = make_int4(0, 0, 0, 0);
  __syncthreads();

  // ---- stage u[b] (row-major + transposed, f16) ----
  const float* ub = u + (size_t)b * NJ * ND;
  for (int idx = tid; idx < NJ * ND; idx += 256) {
    int j = idx / ND, d = idx - j * ND;
    _Float16 hv = (_Float16)ub[idx];
    uh[j * LD_UH + d] = hv;
    uT[d * LD_UT + j] = hv;
  }
  // ---- stage hw = h * w_hu for this 64-row tile (f16) ----
  const float* hb = h + ((size_t)b * NT + t0) * ND;
  for (int idx = tid; idx < 64 * ND; idx += 256) {
    int r = idx / ND, d = idx - r * ND;
    hw[r * LD_HW + d] = (_Float16)(hb[(size_t)r * ND + d] * w_hu[d]);
  }

  // ---- tu[j] = u_j . w_u  (wave-cooperative, coalesced) ----
#pragma unroll 1
  for (int jj = 0; jj < 32; ++jj) {
    int j = w * 32 + jj;
    const float* up = ub + (size_t)j * ND;
    float p = 0.f;
    for (int d = lane; d < ND; d += 32) p += up[d] * w_u[d];
#pragma unroll
    for (int off = 16; off > 0; off >>= 1) p += __shfl_xor(p, off);
    if (lane == 0) tu[j] = p;
  }
  // ---- th[r] = h_r . w_h ----
#pragma unroll 1
  for (int rr = 0; rr < 8; ++rr) {
    int r = w * 8 + rr;
    const float* hp = hb + (size_t)r * ND;
    float p = 0.f;
    for (int d = lane; d < ND; d += 32) p += hp[d] * w_h[d];
#pragma unroll
    for (int off = 16; off > 0; off >>= 1) p += __shfl_xor(p, off);
    if (lane == 0) th[r] = p;
  }
  __syncthreads();

  // ---- GEMM1: S(64x256) = HW(64x224) x U^T  via v_wmma_f32_16x16x32_f16 ----
  const int mtile = w >> 1;    // 0..3  -> M rows mtile*16..+15
  const int nhalf = w & 1;     // 0..1  -> 8 N tiles each
  const v8f vz = {0.f,0.f,0.f,0.f,0.f,0.f,0.f,0.f};
  v8f acc[8];
#pragma unroll
  for (int i = 0; i < 8; ++i) acc[i] = vz;

#pragma unroll
  for (int ks = 0; ks < 7; ++ks) {             // K = 224 (200 + zero pad)
    v16h A = frag_a(hw + mtile * 16 * LD_HW + ks * 32, LD_HW);
#pragma unroll
    for (int i = 0; i < 8; ++i) {
      int n0 = (nhalf * 8 + i) * 16;
      v16h Bm = frag_b(uT + ks * 32 * LD_UT + n0, LD_UT);
      acc[i] = __builtin_amdgcn_wmma_f32_16x16x32_f16(
          false, A, false, Bm, (short)0, acc[i], false, false);
    }
  }
  __syncthreads();   // all uT/hw reads complete before aliasing regions are rewritten

  // ---- s = acc + th + tu + bias (f32 into uT region) ----
  const float bias = b_h[0] + b_u[0] + b_hu[0];
  {
    const int cn = lane & 15;
    const int mb = (lane >> 4) * 8;            // C/D layout: lane<16 -> M=v, else M=8+v
#pragma unroll
    for (int i = 0; i < 8; ++i) {
      int n0 = (nhalf * 8 + i) * 16;
#pragma unroll
      for (int v = 0; v < 8; ++v) {
        int r = mtile * 16 + mb + v;
        sS[r * LD_S + n0 + cn] = acc[i][v] + th[r] + tu[n0 + cn] + bias;
      }
    }
  }
  __syncthreads();

  // ---- row softmax over J=256 (4 lanes per row), P = exp(s-max) in f16 ----
  {
    int r = w * 8 + (lane >> 2);
    int q = lane & 3;
    const float* srow = sS + r * LD_S + q * 64;
    float mx = -3.0e38f;
#pragma unroll 4
    for (int i = 0; i < 64; ++i) mx = fmaxf(mx, srow[i]);
    mx = fmaxf(mx, __shfl_xor(mx, 1));
    mx = fmaxf(mx, __shfl_xor(mx, 2));
    float sm = 0.f;
    _Float16* prow = P + r * LD_P + q * 64;
#pragma unroll 4
    for (int i = 0; i < 64; ++i) {
      float e = __expf(srow[i] - mx);
      prow[i] = (_Float16)e;
      sm += e;
    }
    sm += __shfl_xor(sm, 1);
    sm += __shfl_xor(sm, 2);
    if (q == 0) {
      rs[r] = sm;                               // row sum (normalize at epilogue)
      mrow_g[(size_t)b * NT + t0 + r] = mx;     // max_j s for query-to-context pass
    }
  }
  __syncthreads();

  // ---- GEMM2: C2Q(64x224) = P(64x256) x U(256x224) ----
  v8f acc2[7];
#pragma unroll
  for (int i = 0; i < 7; ++i) acc2[i] = vz;
#pragma unroll
  for (int ks = 0; ks < 8; ++ks) {             // K = 256
    v16h A = frag_a(P + mtile * 16 * LD_P + ks * 32, LD_P);
#pragma unroll
    for (int i = 0; i < 7; ++i) {
      int n0 = (nhalf * 7 + i) * 16;
      v16h Bm = frag_b(uh + ks * 32 * LD_UH + n0, LD_UH);
      acc2[i] = __builtin_amdgcn_wmma_f32_16x16x32_f16(
          false, A, false, Bm, (short)0, acc2[i], false, false);
    }
  }

  // ---- epilogue: normalize and stream g chunks 1 (c2q) and 2 (h*c2q) ----
  {
    const int cn = lane & 15;
    const int mb = (lane >> 4) * 8;
#pragma unroll
    for (int i = 0; i < 7; ++i) {
      int d = (nhalf * 7 + i) * 16 + cn;
      if (d < ND) {
#pragma unroll
        for (int v = 0; v < 8; ++v) {
          int r = mtile * 16 + mb + v;
          float c2q = acc2[i][v] / rs[r];
          size_t row = (size_t)b * NT + t0 + r;
          float hv = h[row * ND + d];
          g[row * NG + ND + d]     = c2q;
          g[row * NG + 2 * ND + d] = hv * c2q;
        }
      }
    }
  }
}

// ======================================================================
// Kernel 2: per-batch softmax stats over T of mrow; also zero q2c (ws poisoned)
// ======================================================================
__global__ __launch_bounds__(256)
void bidaf_qsoft_kernel(const float* __restrict__ mrow, float* __restrict__ ms,
                        float* __restrict__ q2c)
{
  __shared__ float red[256];
  const int b = blockIdx.x, tid = threadIdx.x;
  const float* mr = mrow + (size_t)b * NT;
  float mx = -3.0e38f;
  for (int t = tid; t < NT; t += 256) mx = fmaxf(mx, mr[t]);
  red[tid] = mx; __syncthreads();
  for (int s = 128; s > 0; s >>= 1) {
    if (tid < s) red[tid] = fmaxf(red[tid], red[tid + s]);
    __syncthreads();
  }
  mx = red[0]; __syncthreads();
  float sm = 0.f;
  for (int t = tid; t < NT; t += 256) sm += __expf(mr[t] - mx);
  red[tid] = sm; __syncthreads();
  for (int s = 128; s > 0; s >>= 1) {
    if (tid < s) red[tid] += red[tid + s];
    __syncthreads();
  }
  if (tid == 0) { ms[b * 2] = mx; ms[b * 2 + 1] = red[0]; }
  for (int d = tid; d < ND; d += 256) q2c[b * ND + d] = 0.f;
}

// ======================================================================
// Kernel 3: q2c[b,d] = sum_t softmax_t(m)[t] * h[b,t,d]   (chunked over T)
// grid (NT/256, NB)
// ======================================================================
__global__ __launch_bounds__(256)
void bidaf_q2c_kernel(const float* __restrict__ h, const float* __restrict__ mrow,
                      const float* __restrict__ ms, float* __restrict__ q2c)
{
  __shared__ float wts[256];
  const int b = blockIdx.y, c = blockIdx.x, tid = threadIdx.x;
  const int t0 = c * 256;
  const float mx  = ms[b * 2];
  const float inv = 1.f / ms[b * 2 + 1];
  wts[tid] = __expf(mrow[(size_t)b * NT + t0 + tid] - mx) * inv;
  __syncthreads();
  if (tid < ND) {
    const float* hp = h + ((size_t)b * NT + t0) * ND + tid;
    float a0 = 0.f, a1 = 0.f, a2 = 0.f, a3 = 0.f;
    for (int i = 0; i < 256; i += 4) {
      a0 += wts[i]     * hp[(size_t)(i)     * ND];
      a1 += wts[i + 1] * hp[(size_t)(i + 1) * ND];
      a2 += wts[i + 2] * hp[(size_t)(i + 2) * ND];
      a3 += wts[i + 3] * hp[(size_t)(i + 3) * ND];
    }
    atomicAdd(&q2c[b * ND + tid], (a0 + a1) + (a2 + a3));
  }
}

// ======================================================================
// Kernel 4: stream g chunks 0 (h) and 3 (h*q2c)  — pure bandwidth
// ======================================================================
__global__ __launch_bounds__(256)
void bidaf_g03_kernel(const float* __restrict__ h, const float* __restrict__ q2c,
                      float* __restrict__ g)
{
  const size_t n = (size_t)NB * NT * ND;
  size_t idx = (size_t)blockIdx.x * 256 + threadIdx.x;
  if (idx >= n) return;
  int d = (int)(idx % ND);
  size_t bt = idx / ND;
  int b = (int)(bt / NT);
  float hv = h[idx];
  float qv = q2c[b * ND + d];
  g[bt * NG + d]          = hv;
  g[bt * NG + 3 * ND + d] = hv * qv;
}

// ======================================================================
extern "C" void kernel_launch(void* const* d_in, const int* in_sizes, int n_in,
                              void* d_out, int out_size, void* d_ws, size_t ws_size,
                              hipStream_t stream)
{
  (void)in_sizes; (void)n_in; (void)out_size; (void)ws_size;
  const float* h    = (const float*)d_in[0];
  const float* u    = (const float*)d_in[1];
  const float* w_h  = (const float*)d_in[2];
  const float* b_h  = (const float*)d_in[3];
  const float* w_u  = (const float*)d_in[4];
  const float* b_u  = (const float*)d_in[5];
  const float* w_hu = (const float*)d_in[6];
  const float* b_hu = (const float*)d_in[7];
  float* g = (float*)d_out;

  // workspace layout
  float* mrow = (float*)d_ws;                      // NB*NT
  float* ms   = mrow + (size_t)NB * NT;            // NB*2 (max, sumexp)
  float* q2c  = ms + NB * 2;                       // NB*ND

  // allow >64KB dynamic LDS (host-side attribute; deterministic every call)
  (void)hipFuncSetAttribute((const void*)bidaf_attn_kernel,
                            hipFuncAttributeMaxDynamicSharedMemorySize, SMEM_TOTAL);

  bidaf_attn_kernel<<<dim3(NT / 64, NB), 256, SMEM_TOTAL, stream>>>(
      h, u, w_h, b_h, w_u, b_u, w_hu, b_hu, g, mrow);
  bidaf_qsoft_kernel<<<NB, 256, 0, stream>>>(mrow, ms, q2c);
  bidaf_q2c_kernel<<<dim3(NT / 256, NB), 256, 0, stream>>>(h, mrow, ms, q2c);
  const size_t nelem = (size_t)NB * NT * ND;
  bidaf_g03_kernel<<<(unsigned)((nelem + 255) / 256), 256, 0, stream>>>(h, q2c, g);
}
